// AttentionBlock3D_86930138071201
// MI455X (gfx1250) — matrix operand order repaired and checked by reading, not compiled, not run on hardware
//
#include <hip/hip_runtime.h>
#include <hip/hip_bf16.h>

#define DEV static __device__ __forceinline__

typedef __attribute__((ext_vector_type(16))) __bf16 bf16x16;
typedef __attribute__((ext_vector_type(8)))  __bf16 bf16x8;
typedef __attribute__((ext_vector_type(8)))  float  f32x8;

// Problem constants (from reference setup)
constexpr int B  = 2;
constexpr int C  = 256;
constexpr int S  = 4096;          // 16*16*16 spatial
constexpr int NH = 8;
constexpr int HD = 32;            // head dim
constexpr int G  = 8;             // groupnorm groups
constexpr int CG = C / G;         // 32 channels per group
constexpr float GN_EPS = 1e-5f;
constexpr float SCALE  = 0.17677669529663687f;   // 32^-0.5

// Workspace layout (bytes, all 256B aligned)
constexpr size_t OFF_STATS = 0;                              // B*G*2 f32
constexpr size_t OFF_HT    = 256;                            // h^T bf16 [b][s][c]
constexpr size_t SZ_HT     = (size_t)B * S * C * 2;
constexpr size_t OFF_QW    = OFF_HT + SZ_HT;                 // qkv_w bf16
constexpr size_t SZ_QW     = (size_t)3 * C * C * 2;
constexpr size_t OFF_PW    = OFF_QW + SZ_QW;                 // proj_w bf16
constexpr size_t SZ_PW     = (size_t)C * C * 2;
constexpr size_t OFF_Q     = OFF_PW + SZ_PW;                 // q bf16 [b][h][s][d] (pre-scaled)
constexpr size_t SZ_QKV    = (size_t)B * NH * S * HD * 2;
constexpr size_t OFF_K     = OFF_Q + SZ_QKV;                 // k bf16 [b][h][s][d]
constexpr size_t OFF_V     = OFF_K + SZ_QKV;                 // v bf16 [b][h][d][s]  (transposed)
constexpr size_t OFF_AO    = OFF_V + SZ_QKV;                 // attn out bf16 [b][s][c]

DEV f32x8 wmma_bf16(bf16x16 a, bf16x16 b, f32x8 c) {
  return __builtin_amdgcn_wmma_f32_16x16x32_bf16(false, a, false, b, (short)0, c, false, false);
}

DEV float lane_read(float v, int srcLane) {   // ds_bpermute broadcast
  return __int_as_float(__builtin_amdgcn_ds_bpermute(srcLane << 2, __float_as_int(v)));
}

// ---------------- GroupNorm stats: one block per (b, g) ----------------
__global__ __launch_bounds__(256) void gn_stats_kernel(const float* __restrict__ x,
                                                       float* __restrict__ stats) {
  const int bg = blockIdx.x;                     // 0 .. B*G-1
  const long base = (long)bg * (CG * S);         // group channels are contiguous in [b][c][s]
  float sum = 0.f, sumsq = 0.f;
  for (int i = threadIdx.x; i < CG * S; i += blockDim.x) {
    float v = x[base + i];
    sum += v; sumsq += v * v;
  }
  const int lane = threadIdx.x & 31;
  #pragma unroll
  for (int m = 16; m >= 1; m >>= 1) {
    sum   += lane_read(sum,   lane ^ m);
    sumsq += lane_read(sumsq, lane ^ m);
  }
  __shared__ float rs[8], rss[8];
  const int wid = threadIdx.x >> 5;
  if (lane == 0) { rs[wid] = sum; rss[wid] = sumsq; }
  __syncthreads();
  if (threadIdx.x == 0) {
    float s = 0.f, ss = 0.f;
    #pragma unroll
    for (int i = 0; i < 8; ++i) { s += rs[i]; ss += rss[i]; }
    const float inv = 1.0f / (float)(CG * S);
    float mu  = s * inv;
    float var = ss * inv - mu * mu;
    stats[2 * bg]     = mu;
    stats[2 * bg + 1] = rsqrtf(var + GN_EPS);
  }
}

// ---------------- GroupNorm apply + transpose -> h^T bf16 [b][s][c] ----------------
__global__ void gn_apply_kernel(const float* __restrict__ x, const float* __restrict__ gw,
                                const float* __restrict__ gb, const float* __restrict__ stats,
                                __bf16* __restrict__ hT) {
  const long N = (long)B * C * S;
  for (long i = (long)blockIdx.x * blockDim.x + threadIdx.x; i < N;
       i += (long)gridDim.x * blockDim.x) {
    long b = i / ((long)C * S);
    long r = i - b * (long)C * S;
    int  c = (int)(r / S);
    int  s = (int)(r - (long)c * S);
    int  bg = (int)b * G + c / CG;
    float v = (x[i] - stats[2 * bg]) * stats[2 * bg + 1] * gw[c] + gb[c];
    hT[(b * S + s) * C + c] = (__bf16)v;
  }
}

// ---------------- f32 -> bf16 weight convert ----------------
__global__ void cvt_bf16_kernel(const float* __restrict__ w, __bf16* __restrict__ o, int n) {
  for (int i = blockIdx.x * blockDim.x + threadIdx.x; i < n; i += gridDim.x * blockDim.x)
    o[i] = (__bf16)w[i];
}

// ---------------- QKV GEMM: [768 x 256] x [256 x 4096] per batch ----------------
// One wave: 16 (o) x 64 (s) output strip, K=256 in 8 WMMA steps.
__global__ __launch_bounds__(256) void qkv_kernel(const __bf16* __restrict__ qw,
                                                  const float*  __restrict__ qkvb,
                                                  const __bf16* __restrict__ hT,
                                                  __bf16* __restrict__ qb,
                                                  __bf16* __restrict__ kb,
                                                  __bf16* __restrict__ vb) {
  const int lane = threadIdx.x & 31;
  const int wid  = threadIdx.x >> 5;
  const int job  = blockIdx.x * 8 + wid;         // B * 48 * 64 = 6144 jobs
  const int sg   = job % (S / 64);
  const int ot   = (job / (S / 64)) % (3 * C / 16);
  const int bb   = job / ((S / 64) * (3 * C / 16));
  const int o0   = ot * 16;
  const int s0   = sg * 64;
  const int hw   = lane >> 4;                    // half-wave
  const int l15  = lane & 15;

  const __bf16* arow = qw + (long)(o0 + l15) * C;
  f32x8 acc[4] = {{}, {}, {}, {}};
  for (int k0 = 0; k0 < C; k0 += 32) {
    bf16x8 alo = *(const bf16x8*)(arow + k0 + 16 * hw);
    bf16x8 ahi = *(const bf16x8*)(arow + k0 + 16 * hw + 8);
    bf16x16 A = __builtin_shufflevector(alo, ahi, 0,1,2,3,4,5,6,7,8,9,10,11,12,13,14,15);
    #pragma unroll
    for (int t = 0; t < 4; ++t) {
      const __bf16* brow = hT + ((long)bb * S + s0 + t * 16 + l15) * C + k0 + 16 * hw;
      bf16x16 Bm = *(const bf16x16*)brow;
      acc[t] = wmma_bf16(A, Bm, acc[t]);
    }
  }

  const int which = o0 / C;                      // 0=q 1=k 2=v
  const int co    = o0 % C;
  const int head  = co / HD;
  const int dbase = co % HD;                     // 0 or 16
  float biasj[8];
  #pragma unroll
  for (int j = 0; j < 8; ++j) biasj[j] = qkvb[o0 + j + 8 * hw];

  const long bh = (long)bb * NH + head;
  #pragma unroll
  for (int t = 0; t < 4; ++t) {
    const int s = s0 + t * 16 + l15;
    if (which == 2) {                            // V -> transposed [b][h][d][s]
      #pragma unroll
      for (int j = 0; j < 8; ++j)
        vb[(bh * HD + dbase + j + 8 * hw) * S + s] = (__bf16)(acc[t][j] + biasj[j]);
    } else {                                     // Q/K -> [b][h][s][d], 16B packed store
      bf16x8 pk;
      #pragma unroll
      for (int j = 0; j < 8; ++j) {
        float v = acc[t][j] + biasj[j];
        if (which == 0) v *= SCALE;              // fold softmax scale into Q
        pk[j] = (__bf16)v;
      }
      __bf16* dst = (which == 0 ? qb : kb) + (bh * S + s) * HD + dbase + 8 * hw;
      *(bf16x8*)dst = pk;
    }
  }
}

// ---------------- Flash attention: one wave = 16 queries, stream 4096 keys ----------------
__global__ __launch_bounds__(256) void attn_kernel(const __bf16* __restrict__ qb,
                                                   const __bf16* __restrict__ kb,
                                                   const __bf16* __restrict__ vb,
                                                   __bf16* __restrict__ ao) {
  const int lane = threadIdx.x & 31;
  const int wid  = threadIdx.x >> 5;
  const int job  = blockIdx.x * 8 + wid;         // B*NH*(S/16) = 4096 jobs
  const int qt   = job % (S / 16);
  const int head = (job / (S / 16)) % NH;
  const int bb   = job / ((S / 16) * NH);
  const int q0   = qt * 16;
  const int hw   = lane >> 4;
  const int l15  = lane & 15;
  const long bh  = (long)bb * NH + head;

  // Q as B-operand of S^T = K * Q^T : lane = query, 16 contiguous d's
  const __bf16* qrow = qb + (bh * S + q0 + l15) * HD + 16 * hw;
  const bf16x16 Bq = *(const bf16x16*)qrow;

  const __bf16* kbp = kb + bh * S * HD;
  const __bf16* vbp = vb + bh * HD * S;

  f32x8 o0v = {}, o1v = {};
  const f32x8 zc = {};
  float m_run = -__builtin_inff();
  float l_run = 0.f;

  for (int t0 = 0; t0 < S; t0 += 32) {
    // K-tiles as A operands (16 keys x 32 d each)
    const __bf16* k0r = kbp + (long)(t0 + l15) * HD;
    bf16x16 A0 = __builtin_shufflevector(*(const bf16x8*)(k0r + 16 * hw),
                                         *(const bf16x8*)(k0r + 16 * hw + 8),
                                         0,1,2,3,4,5,6,7,8,9,10,11,12,13,14,15);
    const __bf16* k1r = k0r + (long)16 * HD;
    bf16x16 A1 = __builtin_shufflevector(*(const bf16x8*)(k1r + 16 * hw),
                                         *(const bf16x8*)(k1r + 16 * hw + 8),
                                         0,1,2,3,4,5,6,7,8,9,10,11,12,13,14,15);
    f32x8 s0 = wmma_bf16(A0, Bq, zc);   // S^T[key t0+j+8hw][query l15]
    f32x8 s1 = wmma_bf16(A1, Bq, zc);   // keys +16

    // online softmax per query (row stats live in lane = query, both halves identical)
    float mt = -__builtin_inff();
    #pragma unroll
    for (int j = 0; j < 8; ++j) mt = fmaxf(mt, fmaxf(s0[j], s1[j]));
    mt = fmaxf(mt, lane_read(mt, lane ^ 16));
    const float m_new = fmaxf(m_run, mt);
    const float alpha = __expf(m_run - m_new);
    float p0[8], p1[8], ps = 0.f;
    #pragma unroll
    for (int j = 0; j < 8; ++j) {
      p0[j] = __expf(s0[j] - m_new);
      p1[j] = __expf(s1[j] - m_new);
      ps += p0[j] + p1[j];
    }
    ps += lane_read(ps, lane ^ 16);
    l_run = l_run * alpha + ps;
    m_run = m_new;

    // rescale O accumulators (row m = j + 8hw -> alpha held in lane 8hw+j)
    #pragma unroll
    for (int j = 0; j < 8; ++j) {
      float aj = lane_read(alpha, 8 * hw + j);
      o0v[j] *= aj; o1v[j] *= aj;
    }

    // P already sits in A-operand layout thanks to transposed scores
    bf16x16 Ap;
    #pragma unroll
    for (int i = 0; i < 8; ++i) { Ap[i] = (__bf16)p0[i]; Ap[8 + i] = (__bf16)p1[i]; }

    // V^T B-operands: lane = d, keys t0+8hw..+7 then t0+16+8hw..+7 (matches P)
    const __bf16* v0r = vbp + (long)l15 * S + t0 + 8 * hw;
    const __bf16* v1r = vbp + (long)(16 + l15) * S + t0 + 8 * hw;
    bf16x16 Bv0 = __builtin_shufflevector(*(const bf16x8*)v0r, *(const bf16x8*)(v0r + 16),
                                          0,1,2,3,4,5,6,7,8,9,10,11,12,13,14,15);
    bf16x16 Bv1 = __builtin_shufflevector(*(const bf16x8*)v1r, *(const bf16x8*)(v1r + 16),
                                          0,1,2,3,4,5,6,7,8,9,10,11,12,13,14,15);
    o0v = wmma_bf16(Ap, Bv0, o0v);
    o1v = wmma_bf16(Ap, Bv1, o1v);
  }

  const float linv = 1.0f / l_run;
  #pragma unroll
  for (int j = 0; j < 8; ++j) {
    float lj = lane_read(linv, 8 * hw + j);
    int s = q0 + j + 8 * hw;
    long base = ((long)bb * S + s) * C + head * HD;
    ao[base + l15]      = (__bf16)(o0v[j] * lj);
    ao[base + 16 + l15] = (__bf16)(o1v[j] * lj);
  }
}

// ---------------- Proj GEMM + bias + residual (f32 out) ----------------
__global__ __launch_bounds__(256) void proj_kernel(const __bf16* __restrict__ pw,
                                                   const float*  __restrict__ pb,
                                                   const __bf16* __restrict__ ao,
                                                   const float*  __restrict__ x,
                                                   float* __restrict__ out) {
  const int lane = threadIdx.x & 31;
  const int wid  = threadIdx.x >> 5;
  const int job  = blockIdx.x * 8 + wid;         // B * 16 * 64 = 2048 jobs
  const int sg   = job % (S / 64);
  const int ot   = (job / (S / 64)) % (C / 16);
  const int bb   = job / ((S / 64) * (C / 16));
  const int o0   = ot * 16;
  const int s0   = sg * 64;
  const int hw   = lane >> 4;
  const int l15  = lane & 15;

  const __bf16* arow = pw + (long)(o0 + l15) * C;
  f32x8 acc[4] = {{}, {}, {}, {}};
  for (int k0 = 0; k0 < C; k0 += 32) {
    bf16x8 alo = *(const bf16x8*)(arow + k0 + 16 * hw);
    bf16x8 ahi = *(const bf16x8*)(arow + k0 + 16 * hw + 8);
    bf16x16 A = __builtin_shufflevector(alo, ahi, 0,1,2,3,4,5,6,7,8,9,10,11,12,13,14,15);
    #pragma unroll
    for (int t = 0; t < 4; ++t) {
      const __bf16* brow = ao + ((long)bb * S + s0 + t * 16 + l15) * C + k0 + 16 * hw;
      acc[t] = wmma_bf16(A, *(const bf16x16*)brow, acc[t]);
    }
  }
  float bj[8];
  #pragma unroll
  for (int j = 0; j < 8; ++j) bj[j] = pb[o0 + j + 8 * hw];
  #pragma unroll
  for (int t = 0; t < 4; ++t) {
    const int s = s0 + t * 16 + l15;
    #pragma unroll
    for (int j = 0; j < 8; ++j) {
      const int o = o0 + j + 8 * hw;
      const long xi = ((long)bb * C + o) * S + s;
      out[xi] = acc[t][j] + bj[j] + x[xi];
    }
  }
}

extern "C" void kernel_launch(void* const* d_in, const int* in_sizes, int n_in,
                              void* d_out, int out_size, void* d_ws, size_t ws_size,
                              hipStream_t stream) {
  (void)in_sizes; (void)n_in; (void)out_size; (void)ws_size;
  const float* x    = (const float*)d_in[0];
  const float* gnw  = (const float*)d_in[1];
  const float* gnb  = (const float*)d_in[2];
  const float* qkvw = (const float*)d_in[3];
  const float* qkvb = (const float*)d_in[4];
  const float* pw   = (const float*)d_in[5];
  const float* pb   = (const float*)d_in[6];
  float* out = (float*)d_out;

  char* ws = (char*)d_ws;
  float*  stats = (float*) (ws + OFF_STATS);
  __bf16* hT    = (__bf16*)(ws + OFF_HT);
  __bf16* qwb   = (__bf16*)(ws + OFF_QW);
  __bf16* pwb   = (__bf16*)(ws + OFF_PW);
  __bf16* qb    = (__bf16*)(ws + OFF_Q);
  __bf16* kb    = (__bf16*)(ws + OFF_K);
  __bf16* vb    = (__bf16*)(ws + OFF_V);
  __bf16* ao    = (__bf16*)(ws + OFF_AO);

  gn_stats_kernel<<<B * G, 256, 0, stream>>>(x, stats);
  gn_apply_kernel<<<1024, 256, 0, stream>>>(x, gnw, gnb, stats, hT);
  cvt_bf16_kernel<<<192, 256, 0, stream>>>(qkvw, qwb, 3 * C * C);
  cvt_bf16_kernel<<<64, 256, 0, stream>>>(pw, pwb, C * C);
  qkv_kernel<<<768, 256, 0, stream>>>(qwb, qkvb, hT, qb, kb, vb);
  attn_kernel<<<512, 256, 0, stream>>>(qb, kb, vb, ao);
  proj_kernel<<<256, 256, 0, stream>>>(pwb, pb, ao, x, out);
}
